// BoxRFDGCNN_27754078667218
// MI455X (gfx1250) — compile-verified, matrix operand order checked by reading, not run on hardware
//
#include <hip/hip_runtime.h>
#include <hip/hip_bf16.h>

typedef __attribute__((ext_vector_type(16))) _Float16 v16h;
typedef __attribute__((ext_vector_type(8)))  float    v8f;
typedef __attribute__((ext_vector_type(4)))  float    f32x4;

union V16 { v16h h; f32x4 f[2]; };

#define NPTS 8192
#define KNN  10
#define BNEPS 1e-5f
#define FLT_BIG 3.4e38f

// ---------------------------------------------------------------------------
// Kernel 1: feature embedding branches + fusion.  fused[N,128] f32 + f16 + |x|^2
// ---------------------------------------------------------------------------
__global__ void embed_fuse(const float* __restrict__ xb, const float* __restrict__ xr,
                           const float* __restrict__ xt,
                           const float* __restrict__ Wn, const float* __restrict__ bn_,
                           const float* __restrict__ Wr, const float* __restrict__ br,
                           const float* __restrict__ Wt, const float* __restrict__ bt,
                           const float* __restrict__ Wf, const float* __restrict__ bf,
                           float* __restrict__ fused, _Float16* __restrict__ fusedH,
                           float* __restrict__ sq)
{
    __shared__ float sb[8], sr[64], st[32], scat[192], sred[128];
    const int i = blockIdx.x, t = threadIdx.x;            // 128 threads
    if (t < 8)  sb[t] = xb[(size_t)i * 8  + t];
    if (t < 64) sr[t] = xr[(size_t)i * 64 + t];
    if (t < 32) st[t] = xt[(size_t)i * 32 + t];
    __syncthreads();
    if (t < 64) {
        float a = bn_[t];
        #pragma unroll
        for (int c = 0; c < 8;  ++c) a += sb[c] * Wn[c * 64 + t];
        scat[t] = fmaxf(a, 0.f);
        float r = br[t];
        for (int c = 0; c < 64; ++c) r += sr[c] * Wr[c * 64 + t];
        scat[64 + t] = fmaxf(r, 0.f);
        float x = bt[t];
        for (int c = 0; c < 32; ++c) x += st[c] * Wt[c * 64 + t];
        scat[128 + t] = fmaxf(x, 0.f);
    }
    __syncthreads();
    float f = bf[t];
    for (int c = 0; c < 192; ++c) f += scat[c] * Wf[c * 128 + t];
    f = f > 0.f ? f : 0.01f * f;                          // leaky relu
    fused[(size_t)i * 128 + t]  = f;
    fusedH[(size_t)i * 128 + t] = (_Float16)f;
    sred[t] = f * f;
    __syncthreads();
    for (int s = 64; s > 0; s >>= 1) { if (t < s) sred[t] += sred[t + s]; __syncthreads(); }
    if (t == 0) sq[i] = sred[0];
}

// ---------------------------------------------------------------------------
// Kernel 2: fused kNN via WMMA Gram tiles, async double-buffered B staging in
// LDS, and in-register top-10.  Block = 256 threads = 8 waves; each wave owns
// a 16-row strip; the whole block shares each 32-column B tile via
// global_load_async_to_lds_b128 (ASYNCcnt) -> ds_load_b128 -> v_wmma.
// A/B 16-bit WMMA layout: lane l -> point (l&15); halves [0..7] = K {8*hi..+7},
// halves [8..15] = K {16+8*hi..+7}  => two contiguous 16B reads per 32-K chunk.
// ---------------------------------------------------------------------------
template <int D>
__global__ __launch_bounds__(256) void knn_topk(const _Float16* __restrict__ feats,
                                                const float* __restrict__ sq,
                                                int* __restrict__ knn, int n)
{
    constexpr int NC   = D / 32;                 // K chunks per WMMA row
    constexpr int NLDR = (32 * D * 2) / (16 * 256); // b128 async loads / thread / tile
    __shared__ __align__(16) _Float16 sB[2][32][D];
    __shared__ float sdist[8][16][32];
    __shared__ float smd[8][16][20];
    __shared__ int   smi[8][16][20];

    const int wave = threadIdx.x >> 5, lane = threadIdx.x & 31;
    const int p = lane & 15, hi = lane >> 4;
    const int row0 = blockIdx.x * 128 + wave * 16;

    // A fragments (this wave's 16 rows), loaded once from global
    v16h A[NC];
    const _Float16* arow = feats + (size_t)(row0 + p) * D;
    #pragma unroll
    for (int c = 0; c < NC; ++c) {
        V16 u;
        u.f[0] = *(const f32x4*)(arow + c * 32 + 8 * hi);
        u.f[1] = *(const f32x4*)(arow + c * 32 + 16 + 8 * hi);
        A[c] = u.h;
    }
    float sqa[8];
    #pragma unroll
    for (int r = 0; r < 8; ++r) sqa[r] = sq[row0 + hi * 8 + r];

    float topd[KNN]; int topi[KNN];
    #pragma unroll
    for (int s = 0; s < KNN; ++s) { topd[s] = FLT_BIG; topi[s] = 0; }

    // Stage one 32-point B tile into LDS buffer `buf` via async DMA.
    auto stage = [&](int colbase, int buf) {
        #pragma unroll
        for (int q = 0; q < NLDR; ++q) {
            const int e16 = threadIdx.x + q * 256;       // 16-byte chunk id
            const int row = e16 / (D / 8);
            const int off = (e16 % (D / 8)) * 8;
            const _Float16* gsrc = feats + (size_t)(colbase + row) * D + off;
            const unsigned lds = (unsigned)(size_t)(&sB[buf][row][off]);
            asm volatile("global_load_async_to_lds_b128 %0, %1, off"
                         :: "v"(lds), "v"(gsrc) : "memory");
        }
    };

    const int ntiles = n / 32;
    stage(0, 0);
    for (int it = 0; it < ntiles; ++it) {
        const int colbase = it * 32;
        const int buf = it & 1;
        if (it + 1 < ntiles) {
            stage(colbase + 32, buf ^ 1);                // prefetch next tile
            // async loads complete in order: <=NLDR outstanding => tile `buf` done
            if constexpr (NLDR == 2) asm volatile("s_wait_asynccnt 0x2" ::: "memory");
            else                     asm volatile("s_wait_asynccnt 0x1" ::: "memory");
        } else {
            asm volatile("s_wait_asynccnt 0x0" ::: "memory");
        }
        __syncthreads();                                 // tile visible to all waves

        v8f c0 = {}, c1 = {};
        #pragma unroll
        for (int c = 0; c < NC; ++c) {
            V16 u0, u1;
            u0.f[0] = *(const f32x4*)(&sB[buf][p     ][c * 32 + 8 * hi]);
            u0.f[1] = *(const f32x4*)(&sB[buf][p     ][c * 32 + 16 + 8 * hi]);
            u1.f[0] = *(const f32x4*)(&sB[buf][16 + p][c * 32 + 8 * hi]);
            u1.f[1] = *(const f32x4*)(&sB[buf][16 + p][c * 32 + 16 + 8 * hi]);
            c0 = __builtin_amdgcn_wmma_f32_16x16x32_f16(false, A[c], false, u0.h,
                                                        (short)0, c0, false, false);
            c1 = __builtin_amdgcn_wmma_f32_16x16x32_f16(false, A[c], false, u1.h,
                                                        (short)0, c1, false, false);
        }
        const float sqb0 = sq[colbase + p], sqb1 = sq[colbase + 16 + p];
        #pragma unroll
        for (int r = 0; r < 8; ++r) {
            const int rg = row0 + hi * 8 + r;
            float d0 = sqa[r] + sqb0 - 2.f * c0[r];
            float d1 = sqa[r] + sqb1 - 2.f * c1[r];
            if (rg == colbase + p)       d0 = FLT_BIG;   // exclude self
            if (rg == colbase + 16 + p)  d1 = FLT_BIG;
            sdist[wave][hi * 8 + r][p]      = d0;
            sdist[wave][hi * 8 + r][16 + p] = d1;
        }
        asm volatile("s_wait_dscnt 0" ::: "memory");     // cross-lane LDS visibility
        const int cb = colbase + hi * 16;
        #pragma unroll
        for (int j = 0; j < 16; ++j) {
            float d = sdist[wave][p][hi * 16 + j];
            if (d < topd[KNN - 1]) {
                int id = cb + j;
                #pragma unroll
                for (int s = 0; s < KNN; ++s) {
                    if (d < topd[s]) {
                        float td = topd[s]; int ti = topi[s];
                        topd[s] = d; topi[s] = id; d = td; id = ti;
                    }
                }
            }
        }
        __syncthreads();                                 // reads done before restage
    }
    // merge the two sorted half-lists per row
    #pragma unroll
    for (int s = 0; s < KNN; ++s) { smd[wave][p][hi * KNN + s] = topd[s];
                                    smi[wave][p][hi * KNN + s] = topi[s]; }
    asm volatile("s_wait_dscnt 0" ::: "memory");
    if (hi == 0) {
        int a = 0, b = 0;
        for (int s = 0; s < KNN; ++s) {
            const float da = smd[wave][p][a], db = smd[wave][p][KNN + b];
            int sel;
            if (da <= db) { sel = smi[wave][p][a]; ++a; }
            else          { sel = smi[wave][p][KNN + b]; ++b; }
            knn[(size_t)(row0 + p) * KNN + s] = sel;
        }
    }
}

// ---------------------------------------------------------------------------
// Kernel 3: EdgeConv1  e=[x_i, x_j-x_i] (256) -> 64 -> 64 -> 64, max over K
// ---------------------------------------------------------------------------
__global__ void edge_conv1(const float* __restrict__ fused, const int* __restrict__ knn,
    const float* __restrict__ W1, const float* __restrict__ b1,
    const float* __restrict__ g1, const float* __restrict__ be1,
    const float* __restrict__ m1, const float* __restrict__ v1,
    const float* __restrict__ W2, const float* __restrict__ b2,
    const float* __restrict__ g2, const float* __restrict__ be2,
    const float* __restrict__ m2, const float* __restrict__ v2,
    const float* __restrict__ W3, const float* __restrict__ b3,
    const float* __restrict__ g3, const float* __restrict__ be3,
    const float* __restrict__ m3, const float* __restrict__ v3,
    float* __restrict__ x1, _Float16* __restrict__ x1H, float* __restrict__ sq)
{
    __shared__ float se[256], sh1[64], sh2[64], sred[64];
    const int i = blockIdx.x, t = threadIdx.x;            // 64 threads
    se[t]      = fused[(size_t)i * 128 + t];
    se[64 + t] = fused[(size_t)i * 128 + 64 + t];
    const float s1 = g1[t] * rsqrtf(v1[t] + BNEPS), o1 = be1[t] - m1[t] * s1;
    const float s2 = g2[t] * rsqrtf(v2[t] + BNEPS), o2 = be2[t] - m2[t] * s2;
    const float s3 = g3[t] * rsqrtf(v3[t] + BNEPS), o3 = be3[t] - m3[t] * s3;
    float xm = -FLT_BIG;
    for (int k = 0; k < KNN; ++k) {
        const int j = knn[(size_t)i * KNN + k];
        const float xj0 = fused[(size_t)j * 128 + t];
        const float xj1 = fused[(size_t)j * 128 + 64 + t];
        if (k + 1 < KNN) {                                // hide next gather latency
            const int jn = knn[(size_t)i * KNN + k + 1];
            __builtin_prefetch(&fused[(size_t)jn * 128 + t], 0, 1);
            __builtin_prefetch(&fused[(size_t)jn * 128 + 64 + t], 0, 1);
        }
        __syncthreads();
        se[128 + t] = xj0 - se[t];
        se[192 + t] = xj1 - se[64 + t];
        __syncthreads();
        float a = b1[t];
        for (int c = 0; c < 256; ++c) a += se[c] * W1[c * 64 + t];
        a = a > 0.f ? a : 0.01f * a;  a = a * s1 + o1;
        sh1[t] = a; __syncthreads();
        float h = b2[t];
        for (int c = 0; c < 64; ++c) h += sh1[c] * W2[c * 64 + t];
        h = h > 0.f ? h : 0.01f * h;  h = h * s2 + o2;
        sh2[t] = h; __syncthreads();
        float z = b3[t];
        for (int c = 0; c < 64; ++c) z += sh2[c] * W3[c * 64 + t];
        z = z > 0.f ? z : 0.01f * z;  z = z * s3 + o3;
        z = fminf(fmaxf(z, -10.f), 10.f);
        xm = fmaxf(xm, z);
    }
    x1[(size_t)i * 64 + t]  = xm;
    x1H[(size_t)i * 64 + t] = (_Float16)xm;
    sred[t] = xm * xm; __syncthreads();
    for (int s = 32; s > 0; s >>= 1) { if (t < s) sred[t] += sred[t + s]; __syncthreads(); }
    if (t == 0) sq[i] = sred[0];
}

// ---------------------------------------------------------------------------
// Kernel 4: EdgeConv2  e=[x_i, x_j-x_i] (128) -> 128, max over K
// ---------------------------------------------------------------------------
__global__ void edge_conv2(const float* __restrict__ x1, const int* __restrict__ knn,
    const float* __restrict__ W, const float* __restrict__ b,
    const float* __restrict__ g, const float* __restrict__ be,
    const float* __restrict__ m, const float* __restrict__ v,
    float* __restrict__ x2)
{
    __shared__ float se[128];
    const int i = blockIdx.x, t = threadIdx.x;            // 128 threads
    if (t < 64) se[t] = x1[(size_t)i * 64 + t];
    const float s = g[t] * rsqrtf(v[t] + BNEPS), o = be[t] - m[t] * s;
    float xm = -FLT_BIG;
    for (int k = 0; k < KNN; ++k) {
        const int j = knn[(size_t)i * KNN + k];
        if (k + 1 < KNN && t < 64) {
            const int jn = knn[(size_t)i * KNN + k + 1];
            __builtin_prefetch(&x1[(size_t)jn * 64 + t], 0, 1);
        }
        __syncthreads();
        if (t < 64) se[64 + t] = x1[(size_t)j * 64 + t] - se[t];
        __syncthreads();
        float a = b[t];
        for (int c = 0; c < 128; ++c) a += se[c] * W[c * 128 + t];
        a = a > 0.f ? a : 0.01f * a;  a = a * s + o;
        a = fminf(fmaxf(a, -10.f), 10.f);
        xm = fmaxf(xm, a);
    }
    x2[(size_t)i * 128 + t] = xm;
}

// ---------------------------------------------------------------------------
// Kernel 5: output head  cat[x1,x2] -> BN(192) -> Linear 192->128 -> ReLU
// ---------------------------------------------------------------------------
__global__ void out_head(const float* __restrict__ x1, const float* __restrict__ x2,
    const float* __restrict__ g, const float* __restrict__ be,
    const float* __restrict__ m, const float* __restrict__ v,
    const float* __restrict__ W, const float* __restrict__ b,
    float* __restrict__ out)
{
    __shared__ float sc[192];
    const int i = blockIdx.x, t = threadIdx.x;            // 128 threads
    if (t < 64) {
        const float x = x1[(size_t)i * 64 + t];
        sc[t] = (x - m[t]) * rsqrtf(v[t] + BNEPS) * g[t] + be[t];
    }
    {
        const float x = x2[(size_t)i * 128 + t];
        const int c = 64 + t;
        sc[c] = (x - m[c]) * rsqrtf(v[c] + BNEPS) * g[c] + be[c];
    }
    __syncthreads();
    float a = b[t];
    for (int c = 0; c < 192; ++c) a += sc[c] * W[c * 128 + t];
    out[(size_t)i * 128 + t] = fmaxf(a, 0.f);
}

// ---------------------------------------------------------------------------
extern "C" void kernel_launch(void* const* d_in, const int* in_sizes, int n_in,
                              void* d_out, int out_size, void* d_ws, size_t ws_size,
                              hipStream_t stream)
{
    const float* x_bbox = (const float*)d_in[0];
    const float* x_rf   = (const float*)d_in[1];
    const float* x_txp  = (const float*)d_in[2];
    const float* Wn  = (const float*)d_in[3],  *bn_ = (const float*)d_in[4];
    const float* Wr  = (const float*)d_in[5],  *br  = (const float*)d_in[6];
    const float* Wt  = (const float*)d_in[7],  *bt  = (const float*)d_in[8];
    const float* Wf  = (const float*)d_in[9],  *bf  = (const float*)d_in[10];
    const float* W1  = (const float*)d_in[11], *b1  = (const float*)d_in[12];
    const float* g1  = (const float*)d_in[13], *be1 = (const float*)d_in[14];
    const float* m1  = (const float*)d_in[15], *v1  = (const float*)d_in[16];
    const float* W2  = (const float*)d_in[17], *b2  = (const float*)d_in[18];
    const float* g2  = (const float*)d_in[19], *be2 = (const float*)d_in[20];
    const float* m2  = (const float*)d_in[21], *v2  = (const float*)d_in[22];
    const float* W3  = (const float*)d_in[23], *b3  = (const float*)d_in[24];
    const float* g3  = (const float*)d_in[25], *be3 = (const float*)d_in[26];
    const float* m3  = (const float*)d_in[27], *v3  = (const float*)d_in[28];
    const float* Wc2 = (const float*)d_in[29], *bc2 = (const float*)d_in[30];
    const float* gc2 = (const float*)d_in[31], *bec2= (const float*)d_in[32];
    const float* mc2 = (const float*)d_in[33], *vc2 = (const float*)d_in[34];
    const float* go  = (const float*)d_in[35], *beo = (const float*)d_in[36];
    const float* mo  = (const float*)d_in[37], *vo  = (const float*)d_in[38];
    const float* Wo  = (const float*)d_in[39], *bo  = (const float*)d_in[40];

    char* ws = (char*)d_ws;
    float*    fused  = (float*)ws;     ws += (size_t)NPTS * 128 * sizeof(float);
    _Float16* fusedH = (_Float16*)ws;  ws += (size_t)NPTS * 128 * sizeof(_Float16);
    float*    sq1    = (float*)ws;     ws += (size_t)NPTS * sizeof(float);
    int*      knn1   = (int*)ws;       ws += (size_t)NPTS * KNN * sizeof(int);
    float*    x1     = (float*)ws;     ws += (size_t)NPTS * 64 * sizeof(float);
    _Float16* x1H    = (_Float16*)ws;  ws += (size_t)NPTS * 64 * sizeof(_Float16);
    float*    sq2    = (float*)ws;     ws += (size_t)NPTS * sizeof(float);
    int*      knn2   = (int*)ws;       ws += (size_t)NPTS * KNN * sizeof(int);
    float*    x2     = (float*)ws;     ws += (size_t)NPTS * 128 * sizeof(float);

    embed_fuse<<<NPTS, 128, 0, stream>>>(x_bbox, x_rf, x_txp, Wn, bn_, Wr, br,
                                         Wt, bt, Wf, bf, fused, fusedH, sq1);
    knn_topk<128><<<NPTS / 128, 256, 0, stream>>>(fusedH, sq1, knn1, NPTS);
    edge_conv1<<<NPTS, 64, 0, stream>>>(fused, knn1,
                                        W1, b1, g1, be1, m1, v1,
                                        W2, b2, g2, be2, m2, v2,
                                        W3, b3, g3, be3, m3, v3,
                                        x1, x1H, sq2);
    knn_topk<64><<<NPTS / 128, 256, 0, stream>>>(x1H, sq2, knn2, NPTS);
    edge_conv2<<<NPTS, 128, 0, stream>>>(x1, knn2, Wc2, bc2, gc2, bec2, mc2, vc2, x2);
    out_head<<<NPTS, 128, 0, stream>>>(x1, x2, go, beo, mo, vo, Wo, bo, (float*)d_out);
}
